// CentroidsLoss_p_56942676410583
// MI455X (gfx1250) — matrix operand order repaired and checked by reading, not compiled, not run on hardware
//
#include <hip/hip_runtime.h>
#include <hip/hip_bf16.h>

// ---------------------------------------------------------------------------
// CentroidsLoss on MI455X (gfx1250), wave32 + WMMA bf16.
//   M = B*P = 8192 rows, N = C = 5000 centroids, K = F = 512.
//   loss = mean_{b,p} relu( dist(b, tgt[b], p) - min_{c != tgt[b]} dist(b,c,p) + 0.3 )
// Strategy: bf16 WMMA GEMM for the cross term, f32 row/col norms, running min
// on d^2 (sqrt deferred to the final 32-wide reduction; sqrt is monotonic and
// min(sqrt(max(d2,0))) == sqrt(max(min(d2),0)) exactly).
// ---------------------------------------------------------------------------

typedef __attribute__((ext_vector_type(16))) __bf16 v16bf;
typedef __attribute__((ext_vector_type(8)))  __bf16 v8bf;
typedef __attribute__((ext_vector_type(8)))  float  v8f;

union Frag16 { v16bf v16; v8bf v8[2]; };

#define M_TOT   8192     // B*P
#define N_TOT   5000     // C
#define K_TOT   512      // F
#define P_DIM   8
#define MBLOCK  32       // rows per workgroup
#define NCHUNK  64       // columns per sweep iteration
#define NCHUNKS 79       // ceil(5000/64)
#define MARGIN  0.3f
#define BIGF    3.0e38f

// ---------------- workspace layout (bytes) ----------------
#define WS_OFF_ABF  ((size_t)0)                         // 8192*512*2  = 8388608
#define WS_OFF_BBF  ((size_t)8388608)                   // 5000*512*2  = 5120000
#define WS_OFF_X2   ((size_t)13508608)                  // 8192*4      = 32768
#define WS_OFF_C2   ((size_t)13541376)                  // 5000*4      = 20000
#define WS_OFF_ACC  ((size_t)13561376)                  // 4

// ---------------------------------------------------------------------------
// Kernel 0: zero the scalar accumulator (deterministic per launch).
// ---------------------------------------------------------------------------
__global__ void cl_zero_kernel(float* __restrict__ acc) { acc[0] = 0.0f; }

// ---------------------------------------------------------------------------
// Kernel 1: preprocess.
//   blocks [0, 8192)      : row m of A   -> bf16 row-major [M,K], x2[m]
//   blocks [8192, 13192)  : row c of B   -> bf16 row-major [C,K], c2[c]
// A gather untangles the [B,F,P] layout: A[m][k] = f_p[b, k, p], m = b*8+p.
// ---------------------------------------------------------------------------
__global__ __launch_bounds__(256)
void cl_prep_kernel(const float* __restrict__ f_p,
                    const float* __restrict__ cent,
                    __bf16* __restrict__ Abf,
                    __bf16* __restrict__ Bbf,
                    float*  __restrict__ x2,
                    float*  __restrict__ c2) {
  __shared__ float red[256];
  const int row = blockIdx.x;
  const int t   = threadIdx.x;
  float ss = 0.0f;
  if (row < M_TOT) {
    const int b = row >> 3, p = row & 7;
    const float* src = f_p + (size_t)b * K_TOT * P_DIM + p;
    __bf16* dst = Abf + (size_t)row * K_TOT;
    for (int k = t; k < K_TOT; k += 256) {
      float v = src[(size_t)k * P_DIM];
      dst[k] = (__bf16)v;
      ss += v * v;
    }
  } else {
    const int c = row - M_TOT;
    const float* src = cent + (size_t)c * K_TOT;
    __bf16* dst = Bbf + (size_t)c * K_TOT;
    for (int k = t; k < K_TOT; k += 256) {
      float v = src[k];
      dst[k] = (__bf16)v;
      ss += v * v;
    }
  }
  red[t] = ss;
  __syncthreads();
  for (int s = 128; s > 0; s >>= 1) {
    if (t < s) red[t] += red[t + s];
    __syncthreads();
  }
  if (t == 0) {
    if (row < M_TOT) x2[row] = red[0];
    else             c2[row - M_TOT] = red[0];
  }
}

// ---------------------------------------------------------------------------
// Kernel 2: fused WMMA GEMM + masked-min reduction.
// 256 blocks x 256 threads (8 waves). Block owns MBLOCK=32 rows.
// Wave w: mt = w>>2 (0/1), nt = w&3  -> one 16x16 tile position.
// A fragments (full K=512 for this wave's 16 rows) are register-resident for
// the entire N sweep; the inner loop streams only L2-resident B fragments.
// ---------------------------------------------------------------------------
__global__ __launch_bounds__(256)
void cl_main_kernel(const __bf16* __restrict__ Abf,
                    const __bf16* __restrict__ Bbf,
                    const float* __restrict__ x2,
                    const float* __restrict__ c2,
                    const long long* __restrict__ tgt,
                    float* __restrict__ acc) {
  __shared__ float redmin[MBLOCK][NCHUNK];   // 8 KB: partial d^2 mins
  __shared__ float redpos[MBLOCK][NCHUNK];   // 8 KB: target-class d^2 (max-reduce)

  const int tid    = threadIdx.x;
  const int lane   = tid & 31;
  const int wid    = tid >> 5;               // 0..7
  const int nt     = wid & 3;
  const int mt     = wid >> 2;
  const int laneHi = lane >> 4;              // 0/1
  const int lane15 = lane & 15;
  const int mblock = blockIdx.x * MBLOCK;

  // ---- Hoist A fragments: 16 k-steps x v16bf = 128 VGPRs, loop-invariant ----
  // A 16x32 bf16 layout: lane holds row M = lane&15; elems 0-7 -> K=ka..ka+7,
  // elems 8-15 -> K=ka+16..ka+23, ka = kk*32 + laneHi*8.
  const __bf16* Arow = Abf + (size_t)(mblock + mt * 16 + lane15) * K_TOT;
  Frag16 afr[K_TOT / 32];
#pragma unroll
  for (int kk = 0; kk < K_TOT / 32; ++kk) {
    const int ka = kk * 32 + laneHi * 8;
    afr[kk].v8[0] = *(const v8bf*)(Arow + ka);
    afr[kk].v8[1] = *(const v8bf*)(Arow + ka + 16);
  }

  // Per-element row constants (C/D layout: elem v -> row mt*16 + v + laneHi*8).
  float xr[8];
  int   tg[8];
  float mn[8];                               // running min of d^2 (non-target)
  float pd[8];                               // d^2 at target class (set once)
#pragma unroll
  for (int v = 0; v < 8; ++v) {
    const int mrow = mblock + mt * 16 + v + laneHi * 8;
    xr[v] = x2[mrow];
    tg[v] = (int)tgt[mrow >> 3];             // b = mrow / 8; C < 2^31
    mn[v] = BIGF;
    pd[v] = -1.0f;
  }

  for (int cb = 0; cb < NCHUNKS; ++cb) {
    const int c  = cb * NCHUNK + nt * 16 + lane15;   // this lane's column
    const int cs = (c < N_TOT) ? c : 0;              // clamped for loads
    const __bf16* Brow = Bbf + (size_t)cs * K_TOT;

    v8f accv = {};
#pragma unroll
    for (int kk = 0; kk < K_TOT / 32; ++kk) {
      // B 32x16 bf16 layout: lane holds col N = lane&15; elems 0-7 -> K=kb..kb+7,
      // elems 8-15 -> K=kb+16..kb+23, kb = kk*32 + laneHi*8.
      Frag16 b;
      const int kb = kk * 32 + laneHi * 8;
      b.v8[0] = *(const v8bf*)(Brow + kb);
      b.v8[1] = *(const v8bf*)(Brow + kb + 16);
      accv = __builtin_amdgcn_wmma_f32_16x16x32_bf16(
          /*neg_a=*/false, afr[kk].v16, /*neg_b=*/false, b.v16,
          /*c_mod=*/(short)0, accv, /*reuse_a=*/false, /*reuse_b=*/false);
    }

    // Branchless epilogue on d^2 (no sqrt, no exec-mask churn in the loop).
    const bool  valid = (c < N_TOT);
    const float cc    = c2[cs];
#pragma unroll
    for (int v = 0; v < 8; ++v) {
      const float d2  = xr[v] + cc - 2.0f * accv[v];
      const bool  isT = (c == tg[v]);        // tg < N_TOT, so isT implies valid
      mn[v] = fminf(mn[v], (valid && !isT) ? d2 : BIGF);
      pd[v] = isT ? d2 : pd[v];
    }
  }

  // Cross-lane / cross-wave reduction: 64 partials per row.
#pragma unroll
  for (int v = 0; v < 8; ++v) {
    const int r = mt * 16 + v + laneHi * 8;
    const int s = nt * 16 + lane15;
    redmin[r][s] = mn[v];
    redpos[r][s] = pd[v];
  }
  __syncthreads();

  if (tid < MBLOCK) {
    float m = BIGF, p = -1.0f;
#pragma unroll 8
    for (int s = 0; s < NCHUNK; ++s) {
      m = fminf(m, redmin[tid][s]);
      p = fmaxf(p, redpos[tid][s]);
    }
    // Only 2 sqrts per row, after the reductions (monotonicity of sqrt).
    const float pos  = __builtin_amdgcn_sqrtf(fmaxf(p, 0.0f));
    const float mind = __builtin_amdgcn_sqrtf(fmaxf(m, 0.0f));
    const float l = fmaxf(pos - mind + MARGIN, 0.0f);
    atomicAdd(acc, l);
  }
}

// ---------------------------------------------------------------------------
// Kernel 3: finalize mean.
// ---------------------------------------------------------------------------
__global__ void cl_final_kernel(const float* __restrict__ acc, float* __restrict__ out) {
  out[0] = acc[0] * (1.0f / (float)M_TOT);
}

// ---------------------------------------------------------------------------
extern "C" void kernel_launch(void* const* d_in, const int* in_sizes, int n_in,
                              void* d_out, int out_size, void* d_ws, size_t ws_size,
                              hipStream_t stream) {
  const float*     f_p  = (const float*)d_in[0];       // [1024, 512, 8] f32
  const long long* tgt  = (const long long*)d_in[1];   // [1024] i64
  const float*     cent = (const float*)d_in[2];       // [5000, 512] f32
  float* out = (float*)d_out;

  char* ws = (char*)d_ws;
  __bf16* Abf = (__bf16*)(ws + WS_OFF_ABF);
  __bf16* Bbf = (__bf16*)(ws + WS_OFF_BBF);
  float*  x2  = (float*)(ws + WS_OFF_X2);
  float*  c2  = (float*)(ws + WS_OFF_C2);
  float*  acc = (float*)(ws + WS_OFF_ACC);

  cl_zero_kernel<<<1, 1, 0, stream>>>(acc);
  cl_prep_kernel<<<M_TOT + N_TOT, 256, 0, stream>>>(f_p, cent, Abf, Bbf, x2, c2);
  cl_main_kernel<<<M_TOT / MBLOCK, 256, 0, stream>>>(Abf, Bbf, x2, c2, tgt, acc);
  cl_final_kernel<<<1, 1, 0, stream>>>(acc, out);
}
